// LanguageModel_65678639890956
// MI455X (gfx1250) — matrix operand order repaired and checked by reading, not compiled, not run on hardware
//
#include <hip/hip_runtime.h>

// ---------------------------------------------------------------------------
// Types for CDNA5 WMMA (wave32): v_wmma_f32_16x16x32_bf16
// ---------------------------------------------------------------------------
typedef __attribute__((ext_vector_type(16))) __bf16 v16bf;
typedef __attribute__((ext_vector_type(8)))  __bf16 v8bf;
typedef __attribute__((ext_vector_type(8)))  float  v8f;
typedef int v4i_vs __attribute__((vector_size(16)));   // pointee type of async-LDS builtin

union FragBF16 { v16bf v; v8bf h2[2]; };

#define WMMA_BF16(A, B, C) \
  __builtin_amdgcn_wmma_f32_16x16x32_bf16(false, (A), false, (B), (short)0, (C), false, false)

// Model constants (reference: V=32000, BLK=1024, D=768, H=12, L=4, B=2, T=1024)
#define MB    2
#define MT    1024
#define MD    768
#define MH    12
#define ML    4
#define MV    32000
#define MROWS (MB * MT)      // 2048
#define MFF   (4 * MD)       // 3072
#define HD    64             // head dim

// ---------------------------------------------------------------------------
// CDNA5 async global->LDS copy (ASYNCcnt-tracked), 16 bytes per lane.
// Builtin when available; inline-asm fallback per CDNA5 ISA.
// ---------------------------------------------------------------------------
__device__ __forceinline__ void async_cp16(const __bf16* g, __bf16* l) {
#if __has_builtin(__builtin_amdgcn_global_load_async_to_lds_b128)
  __builtin_amdgcn_global_load_async_to_lds_b128(
      (__attribute__((address_space(1))) v4i_vs*)g,
      (__attribute__((address_space(3))) v4i_vs*)l, 0, 0);
#else
  unsigned ldsOff = (unsigned)(size_t)(__attribute__((address_space(3))) char*)l;
  asm volatile("global_load_async_to_lds_b128 %0, %1, off"
               :: "v"(ldsOff), "v"((unsigned long long)(size_t)g) : "memory");
#endif
}

template <int N>
__device__ __forceinline__ void wait_asynccnt() {
#if __has_builtin(__builtin_amdgcn_s_wait_asynccnt)
  __builtin_amdgcn_s_wait_asynccnt(N);
#else
  asm volatile("s_wait_asynccnt %0" :: "n"(N));
#endif
}

// ---------------------------------------------------------------------------
// Wave reductions via ds_swizzle (group-of-32 XOR patterns; wave32)
// ---------------------------------------------------------------------------
__device__ __forceinline__ float redmax16(float v) {   // across each 16-lane half
  v = fmaxf(v, __int_as_float(__builtin_amdgcn_ds_swizzle(__float_as_int(v), 0x041F)));
  v = fmaxf(v, __int_as_float(__builtin_amdgcn_ds_swizzle(__float_as_int(v), 0x081F)));
  v = fmaxf(v, __int_as_float(__builtin_amdgcn_ds_swizzle(__float_as_int(v), 0x101F)));
  v = fmaxf(v, __int_as_float(__builtin_amdgcn_ds_swizzle(__float_as_int(v), 0x201F)));
  return v;
}
__device__ __forceinline__ float redsum16(float v) {
  v += __int_as_float(__builtin_amdgcn_ds_swizzle(__float_as_int(v), 0x041F));
  v += __int_as_float(__builtin_amdgcn_ds_swizzle(__float_as_int(v), 0x081F));
  v += __int_as_float(__builtin_amdgcn_ds_swizzle(__float_as_int(v), 0x101F));
  v += __int_as_float(__builtin_amdgcn_ds_swizzle(__float_as_int(v), 0x201F));
  return v;
}
__device__ __forceinline__ float redsum32(float v) {
  v = redsum16(v);
  v += __int_as_float(__builtin_amdgcn_ds_swizzle(__float_as_int(v), 0x401F));
  return v;
}

// ---------------------------------------------------------------------------
// x[b,t,:] = tok_emb[idx[b,t],:] + pos_emb[t,:]
// ---------------------------------------------------------------------------
__global__ __launch_bounds__(256) void embed_kernel(
    const int* __restrict__ idx, const float* __restrict__ tok,
    const float* __restrict__ pos, float* __restrict__ x)
{
  int i = blockIdx.x * 256 + threadIdx.x;      // over MROWS*MD
  int r = i / MD, c = i - r * MD;
  int t = r & (MT - 1);
  x[i] = tok[(size_t)idx[r] * MD + c] + pos[(size_t)t * MD + c];
}

// Wt[n*K + k] = bf16(W[k*N + n])   (transpose + downconvert)
__global__ __launch_bounds__(256) void wconv_kernel(
    const float* __restrict__ W, __bf16* __restrict__ Wt, int K, int N)
{
  int i = blockIdx.x * 256 + threadIdx.x;      // over N*K
  int n = i / K, k = i - n * K;
  Wt[i] = (__bf16)W[(size_t)k * N + n];
}

__global__ __launch_bounds__(256) void f2b_kernel(
    const float* __restrict__ src, __bf16* __restrict__ dst)
{
  int i = blockIdx.x * 256 + threadIdx.x;
  dst[i] = (__bf16)src[i];
}

// vT[((b*H+h)*64 + d)*T + t] = v[(b*T+t)*D + h*64 + d]
__global__ __launch_bounds__(256) void vtrans_kernel(
    const __bf16* __restrict__ v, __bf16* __restrict__ vT)
{
  int i = blockIdx.x * 256 + threadIdx.x;      // over B*H*HD*T
  int t  = i & (MT - 1);
  int d  = (i >> 10) & (HD - 1);
  int bh = i >> 16;
  int b  = bh / MH, h = bh - b * MH;
  vT[i] = v[((size_t)(b * MT + t)) * MD + h * HD + d];
}

// ---------------------------------------------------------------------------
// LayerNorm: one row (768) per 256-thread block, bf16 output
// ---------------------------------------------------------------------------
__global__ __launch_bounds__(256) void ln_kernel(
    const float* __restrict__ x, const float* __restrict__ g,
    const float* __restrict__ bta, __bf16* __restrict__ out)
{
  int row = blockIdx.x;
  int c = threadIdx.x, lane = c & 31, w = c >> 5;
  const float* xr = x + (size_t)row * MD;
  float v0 = xr[c], v1 = xr[c + 256], v2 = xr[c + 512];

  __shared__ float part[8];
  __shared__ float mv[2];

  float s = redsum32(v0 + v1 + v2);
  if (lane == 0) part[w] = s;
  __syncthreads();
  if (c == 0) { float t = 0.f; for (int i = 0; i < 8; ++i) t += part[i]; mv[0] = t * (1.f / MD); }
  __syncthreads();
  float mean = mv[0];
  float d0 = v0 - mean, d1 = v1 - mean, d2 = v2 - mean;
  float ss = redsum32(d0 * d0 + d1 * d1 + d2 * d2);
  if (lane == 0) part[w] = ss;
  __syncthreads();
  if (c == 0) { float t = 0.f; for (int i = 0; i < 8; ++i) t += part[i]; mv[1] = rsqrtf(t * (1.f / MD) + 1e-5f); }
  __syncthreads();
  float rstd = mv[1];
  __bf16* orow = out + (size_t)row * MD;
  orow[c]       = (__bf16)(d0 * rstd * g[c]       + bta[c]);
  orow[c + 256] = (__bf16)(d1 * rstd * g[c + 256] + bta[c + 256]);
  orow[c + 512] = (__bf16)(d2 * rstd * g[c + 512] + bta[c + 512]);
}

// ---------------------------------------------------------------------------
// GEMM: C(MxN) = A(MxK bf16) @ Bt(NxK bf16, pre-transposed) + epilogues.
// 256 threads = 8 waves (4 M x 2 N), block tile 128x128, wave tile 32x64.
// K chunks of 32 double-buffered in LDS via async global->LDS (ASYNCcnt).
// Requires M%128==0, N%128==0, K%32==0 (all shapes here satisfy this).
// ---------------------------------------------------------------------------
__global__ __launch_bounds__(256) void gemm_bf16(
    const __bf16* __restrict__ A, const __bf16* __restrict__ Bt,
    const float* __restrict__ bias, const float* __restrict__ resid,
    float* __restrict__ Cf, __bf16* __restrict__ Cb,
    int M, int N, int K, int relu)
{
  __shared__ __align__(16) __bf16 lA[2][128 * 32];   // 8 KB x2
  __shared__ __align__(16) __bf16 lB[2][128 * 32];   // 8 KB x2

  const int tid = threadIdx.x;
  const int wave = tid >> 5;
  const int lane = tid & 31;
  const int wm = wave >> 1, wn = wave & 1;
  const int rowBase = blockIdx.y * 128;
  const int colBase = blockIdx.x * 128;
  const int mrow = lane & 15;
  const int hi = lane >> 4;

  // async fill: 512 16B pieces per tile, 2 per thread per tile
  auto issue_chunk = [&](int k0, int buf) {
#pragma unroll
    for (int j = 0; j < 2; ++j) {
      int idx = tid * 2 + j;
      int r = idx >> 2, part = idx & 3;
      async_cp16(A  + (size_t)(rowBase + r) * K + k0 + part * 8, &lA[buf][r * 32 + part * 8]);
      async_cp16(Bt + (size_t)(colBase + r) * K + k0 + part * 8, &lB[buf][r * 32 + part * 8]);
    }
  };

  v8f acc[2][4] = {};
  const int nk = K / 32;

  issue_chunk(0, 0);
  if (nk > 1) issue_chunk(32, 1);

  for (int kc = 0; kc < nk; ++kc) {
    // retire the chunk we are about to consume; keep the next one in flight
    if (kc + 1 < nk) wait_asynccnt<4>(); else wait_asynccnt<0>();
    __syncthreads();

    const int buf = kc & 1;
    const __bf16* la = &lA[buf][0];
    const __bf16* lb = &lB[buf][0];

    FragBF16 a[2];
#pragma unroll
    for (int mi = 0; mi < 2; ++mi) {
      const __bf16* ar = la + (wm * 32 + mi * 16 + mrow) * 32 + (hi ? 8 : 0);
      a[mi].h2[0] = *(const v8bf*)(ar);
      a[mi].h2[1] = *(const v8bf*)(ar + 16);
    }
#pragma unroll
    for (int j = 0; j < 4; ++j) {
      const __bf16* br = lb + (wn * 64 + j * 16 + mrow) * 32 + hi * 8;
      FragBF16 b;
      b.h2[0] = *(const v8bf*)(br);
      b.h2[1] = *(const v8bf*)(br + 16);
      acc[0][j] = WMMA_BF16(a[0].v, b.v, acc[0][j]);
      acc[1][j] = WMMA_BF16(a[1].v, b.v, acc[1][j]);
    }

    __syncthreads();                 // all waves done reading buf before refill
    if (kc + 2 < nk) issue_chunk((kc + 2) * 32, buf);
  }

#pragma unroll
  for (int mi = 0; mi < 2; ++mi) {
#pragma unroll
    for (int j = 0; j < 4; ++j) {
      const int col = colBase + wn * 64 + j * 16 + mrow;
      const float bv = bias ? bias[col] : 0.f;
#pragma unroll
      for (int r = 0; r < 8; ++r) {
        const int row = rowBase + wm * 32 + mi * 16 + hi * 8 + r;
        float cv = acc[mi][j][r] + bv;
        if (relu) cv = fmaxf(cv, 0.f);
        const size_t off = (size_t)row * N + col;
        if (resid) cv += resid[off];
        if (Cf) Cf[off] = cv;
        if (Cb) Cb[off] = (__bf16)cv;
      }
    }
  }
}

// ---------------------------------------------------------------------------
// Flash attention: one wave per (b, h, 16-query tile). bf16 WMMA for QK^T and
// PV, fp32 online softmax (scale = D^-0.5 per reference), causal mask.
// ---------------------------------------------------------------------------
__global__ __launch_bounds__(32) void attn_kernel(
    const __bf16* __restrict__ q, const __bf16* __restrict__ k,
    const __bf16* __restrict__ vT, __bf16* __restrict__ o)
{
  const int lane = threadIdx.x;
  const int mrow = lane & 15;
  const int hi = lane >> 4;
  const int qt = blockIdx.x & 63;
  const int bh = blockIdx.x >> 6;
  const int b = bh / MH, h = bh - b * MH;

  const __bf16* qr = q + ((size_t)(b * MT + qt * 16 + mrow)) * MD + h * HD + hi * 8;
  FragBF16 aq0, aq1;
  aq0.h2[0] = *(const v8bf*)(qr);
  aq0.h2[1] = *(const v8bf*)(qr + 16);
  aq1.h2[0] = *(const v8bf*)(qr + 32);
  aq1.h2[1] = *(const v8bf*)(qr + 48);

  v8f oacc[4] = {};
  float mrun[8], lrun[8];
#pragma unroll
  for (int r = 0; r < 8; ++r) { mrun[r] = -3.0e38f; lrun[r] = 0.f; }

  __shared__ __align__(32) __bf16 pbuf[16 * 32];

  const __bf16* kb  = k + (size_t)(b * MT) * MD + h * HD;
  const __bf16* vtb = vT + (size_t)bh * HD * MT;
  const float scale = 0.03608439182435161f;   // 768^-0.5 (full embed dim, per ref)
  const int qlimit = qt * 16 + 15;

  for (int k0 = 0; k0 <= qlimit; k0 += 32) {
    float st[2][8];
#pragma unroll
    for (int t = 0; t < 2; ++t) {
      const int key = k0 + t * 16 + mrow;
      const __bf16* kr = kb + (size_t)key * MD + hi * 8;
      FragBF16 blo, bhi;
      blo.h2[0] = *(const v8bf*)(kr);        // head-dims 0..31
      blo.h2[1] = *(const v8bf*)(kr + 16);
      bhi.h2[0] = *(const v8bf*)(kr + 32);   // head-dims 32..63
      bhi.h2[1] = *(const v8bf*)(kr + 48);
      v8f s = {};
      s = WMMA_BF16(aq0.v, blo.v, s);
      s = WMMA_BF16(aq1.v, bhi.v, s);
#pragma unroll
      for (int r = 0; r < 8; ++r) {
        const int qrow = qt * 16 + hi * 8 + r;
        float sv = s[r] * scale;
        if (key > qrow) sv = -3.0e38f;       // causal mask
        st[t][r] = sv;
      }
    }
#pragma unroll
    for (int r = 0; r < 8; ++r) {
      float mx = redmax16(fmaxf(st[0][r], st[1][r]));
      float mnew = fmaxf(mrun[r], mx);
      float alpha = __expf(mrun[r] - mnew);
      float p0 = __expf(st[0][r] - mnew);
      float p1 = __expf(st[1][r] - mnew);
      float rs = redsum16(p0 + p1);
      lrun[r] = lrun[r] * alpha + rs;
      mrun[r] = mnew;
      pbuf[(hi * 8 + r) * 32 + mrow]      = (__bf16)p0;
      pbuf[(hi * 8 + r) * 32 + 16 + mrow] = (__bf16)p1;
      oacc[0][r] *= alpha; oacc[1][r] *= alpha;
      oacc[2][r] *= alpha; oacc[3][r] *= alpha;
    }
    asm volatile("s_wait_dscnt 0x0" ::: "memory");   // all lanes' P visible in LDS
    FragBF16 ap;
    const __bf16* pr = pbuf + mrow * 32 + (hi ? 8 : 0);
    ap.h2[0] = *(const v8bf*)(pr);
    ap.h2[1] = *(const v8bf*)(pr + 16);
#pragma unroll
    for (int j = 0; j < 4; ++j) {
      const __bf16* vr = vtb + (size_t)(j * 16 + mrow) * MT + k0 + hi * 8;
      FragBF16 bv;
      bv.h2[0] = *(const v8bf*)(vr);
      bv.h2[1] = *(const v8bf*)(vr + 16);
      oacc[j] = WMMA_BF16(ap.v, bv.v, oacc[j]);
    }
    asm volatile("s_wait_dscnt 0x0" ::: "memory");   // WAR guard before next chunk
  }

  __bf16* ob = o + ((size_t)(b * MT + qt * 16)) * MD + h * HD;
#pragma unroll
  for (int r = 0; r < 8; ++r) {
    const float inv = 1.0f / lrun[r];
    const int row = hi * 8 + r;
#pragma unroll
    for (int j = 0; j < 4; ++j)
      ob[(size_t)row * MD + j * 16 + mrow] = (__bf16)(oacc[j][r] * inv);
  }
}

// ---------------------------------------------------------------------------
// Host-side orchestration
// ---------------------------------------------------------------------------
static inline void launch_gemm(hipStream_t s,
    const __bf16* A, const __bf16* Bt, const float* bias, const float* resid,
    float* Cf, __bf16* Cb, int M, int N, int K, int relu)
{
  dim3 grid(N / 128, M / 128);
  gemm_bf16<<<grid, 256, 0, s>>>(A, Bt, bias, resid, Cf, Cb, M, N, K, relu);
}

extern "C" void kernel_launch(void* const* d_in, const int* in_sizes, int n_in,
                              void* d_out, int out_size, void* d_ws, size_t ws_size,
                              hipStream_t stream) {
  const int*   idx  = (const int*)  d_in[0];
  const float* tok  = (const float*)d_in[1];
  const float* pos  = (const float*)d_in[2];
  const float* Wq   = (const float*)d_in[3];
  const float* Wk   = (const float*)d_in[4];
  const float* Wv   = (const float*)d_in[5];
  const float* Wo   = (const float*)d_in[6];
  const float* bo   = (const float*)d_in[7];
  const float* ln1g = (const float*)d_in[8];
  const float* ln1b = (const float*)d_in[9];
  const float* ln2g = (const float*)d_in[10];
  const float* ln2b = (const float*)d_in[11];
  const float* W1   = (const float*)d_in[12];
  const float* b1   = (const float*)d_in[13];
  const float* W2   = (const float*)d_in[14];
  const float* b2   = (const float*)d_in[15];
  const float* fW1  = (const float*)d_in[16];
  const float* fb1  = (const float*)d_in[17];
  const float* fW2  = (const float*)d_in[18];
  const float* fb2  = (const float*)d_in[19];
  const float* outW = (const float*)d_in[20];
  const float* outB = (const float*)d_in[21];

  char* w = (char*)d_ws;
  float*  xbuf = (float*) w; w += (size_t)MROWS * MD * 4;   // residual stream, f32
  __bf16* hbuf = (__bf16*)w; w += (size_t)MROWS * MD * 2;   // LN out / bf16 act
  __bf16* qb   = (__bf16*)w; w += (size_t)MROWS * MD * 2;
  __bf16* kbuf = (__bf16*)w; w += (size_t)MROWS * MD * 2;
  __bf16* vbuf = (__bf16*)w; w += (size_t)MROWS * MD * 2;
  __bf16* obuf = (__bf16*)w; w += (size_t)MROWS * MD * 2;
  __bf16* vTb  = (__bf16*)w; w += (size_t)MROWS * MD * 2;   // == B*H*64*T
  __bf16* h1   = (__bf16*)w; w += (size_t)MROWS * MFF * 2;  // MLP intermediate
  __bf16* wt   = (__bf16*)w;                                // weight slab (max V*D)

  const int ELT = MROWS * MD;
  const int NB  = ELT / 256;

  embed_kernel<<<NB, 256, 0, stream>>>(idx, tok, pos, xbuf);

  for (int l = 0; l < ML; ++l) {
    const size_t oDD = (size_t)l * MD * MD;
    const size_t oDF = (size_t)l * MD * MFF;
    // --- attention ---
    ln_kernel<<<MROWS, 256, 0, stream>>>(xbuf, ln1g + (size_t)l * MD, ln1b + (size_t)l * MD, hbuf);

    wconv_kernel<<<(MD * MD) / 256, 256, 0, stream>>>(Wq + oDD, wt, MD, MD);
    launch_gemm(stream, hbuf, wt, nullptr, nullptr, nullptr, qb, MROWS, MD, MD, 0);
    wconv_kernel<<<(MD * MD) / 256, 256, 0, stream>>>(Wk + oDD, wt, MD, MD);
    launch_gemm(stream, hbuf, wt, nullptr, nullptr, nullptr, kbuf, MROWS, MD, MD, 0);
    wconv_kernel<<<(MD * MD) / 256, 256, 0, stream>>>(Wv + oDD, wt, MD, MD);
    launch_gemm(stream, hbuf, wt, nullptr, nullptr, nullptr, vbuf, MROWS, MD, MD, 0);

    vtrans_kernel<<<NB, 256, 0, stream>>>(vbuf, vTb);
    attn_kernel<<<MB * MH * (MT / 16), 32, 0, stream>>>(qb, kbuf, vTb, obuf);

    wconv_kernel<<<(MD * MD) / 256, 256, 0, stream>>>(Wo + oDD, wt, MD, MD);
    launch_gemm(stream, obuf, wt, bo + (size_t)l * MD, xbuf, xbuf, nullptr, MROWS, MD, MD, 0);

    // --- MLP ---
    ln_kernel<<<MROWS, 256, 0, stream>>>(xbuf, ln2g + (size_t)l * MD, ln2b + (size_t)l * MD, hbuf);
    wconv_kernel<<<(MD * MFF) / 256, 256, 0, stream>>>(W1 + oDF, wt, MD, MFF);
    launch_gemm(stream, hbuf, wt, b1 + (size_t)l * MFF, nullptr, nullptr, h1, MROWS, MFF, MD, 1);
    wconv_kernel<<<(MD * MFF) / 256, 256, 0, stream>>>(W2 + oDF, wt, MFF, MD);
    launch_gemm(stream, h1, wt, b2 + (size_t)l * MD, xbuf, xbuf, nullptr, MROWS, MD, MFF, 0);
  }

  // Final (non-residual) FeedForward
  f2b_kernel<<<NB, 256, 0, stream>>>(xbuf, hbuf);
  wconv_kernel<<<(MD * MFF) / 256, 256, 0, stream>>>(fW1, wt, MD, MFF);
  launch_gemm(stream, hbuf, wt, fb1, nullptr, nullptr, h1, MROWS, MFF, MD, 1);
  wconv_kernel<<<(MD * MFF) / 256, 256, 0, stream>>>(fW2, wt, MFF, MD);
  launch_gemm(stream, h1, wt, fb2, nullptr, nullptr, hbuf, MROWS, MD, MFF, 0);

  // Output head -> logits (f32)
  wconv_kernel<<<(MD * MV) / 256, 256, 0, stream>>>(outW, wt, MD, MV);
  launch_gemm(stream, hbuf, wt, outB, nullptr, (float*)d_out, nullptr, MROWS, MV, MD, 0);
}
